// FlowHead_64914135712204
// MI455X (gfx1250) — matrix operand order repaired and checked by reading, not compiled, not run on hardware
//
#include <hip/hip_runtime.h>
#include <hip/hip_bf16.h>
#include <math.h>

typedef __bf16 bf16_t;
typedef __attribute__((ext_vector_type(16))) __bf16 v16bf;
typedef __attribute__((ext_vector_type(8)))  __bf16 v8bf;
typedef __attribute__((ext_vector_type(8)))  float  v8f;

#define FD      8
#define CIN     256
#define HLR     64
#define WLR     64
#define KTOT    (CIN * 9)     // 2304 = 72 k-tiles of 32
#define KCHUNK  576           // K staged per LDS round (2304/576 = 4, 576/32 = 18)
#define KSTRIDE 584           // padded LDS row stride (bank spread, 16B aligned)
#define MPB     32            // pixels (M) per block = 2 sub-tiles of 16
#define NBATCH  16
#define HF      512
#define WF      512

// ---------------------------------------------------------------------------
// Pack OIHW fp32 weights straight into the WMMA B-fragment layout:
//   Bp[ ((kTile*ncTiles + nTile)*32 + lane)*16 + elem ]
//   lane = ((k>>4)&1)*16 + (c&15), elem = k&15  (lanes 0-15: K 0..15, 16-31: K 16..31)
// k = ci*9 + ky*3 + kx; EqualConv scale folded in; c >= CoutReal zero-padded.
// ---------------------------------------------------------------------------
__global__ void pack_weights_frag(const float* __restrict__ w, bf16_t* __restrict__ Bp,
                                  int CoutReal, int CoutPad, float scale) {
    int idx = blockIdx.x * blockDim.x + threadIdx.x;
    int total = KTOT * CoutPad;
    if (idx >= total) return;
    int k = idx / CoutPad, c = idx % CoutPad;
    float v = 0.f;
    if (c < CoutReal) {
        int ci = k / 9, r = k % 9;
        v = w[((c * CIN + ci) * 3 + r / 3) * 3 + (r % 3)] * scale;
    }
    int ncTiles = CoutPad >> 4;
    int lane = ((k >> 4) & 1) * 16 + (c & 15);
    int pos  = (((k >> 5) * ncTiles + (c >> 4)) * 32 + lane) * 16 + (k & 15);
    Bp[pos] = (bf16_t)v;
}

__global__ void f32_to_bf16(const float* __restrict__ x, bf16_t* __restrict__ y, int n) {
    int i = blockIdx.x * blockDim.x + threadIdx.x;
    if (i < n) y[i] = (bf16_t)x[i];
}

// ---------------------------------------------------------------------------
// Implicit-GEMM 3x3 conv (pad 1) via WMMA bf16.
// Block = 256 threads = 8 waves; block owns a 32-pixel M-tile (2 WMMA
// sub-tiles) and up to 8 x 16 output channels. A staged in LDS in 576-K
// chunks; each B fragment (one 32B load/lane) feeds 2 WMMAs.
// ---------------------------------------------------------------------------
__global__ void __launch_bounds__(256)
conv3x3_wmma(const bf16_t* __restrict__ xbf,    // (N, CIN, 64, 64) bf16
             const bf16_t* __restrict__ Bpack,  // fragment-packed weights
             const float*  __restrict__ bias,   // (CoutReal)
             float* __restrict__ out_f32,       // NCHW fp32 (if !outBf)
             bf16_t* __restrict__ out_bf,       // NCHW bf16 (if outBf)
             int CoutPad, int CoutReal, int relu, int outBf)
{
    __shared__ __align__(32) bf16_t Als[MPB * KSTRIDE];   // ~36.5 KB

    const int wave = threadIdx.x >> 5;
    const int lane = threadIdx.x & 31;
    const int ncTiles  = CoutPad >> 4;
    const int ncBlocks = (CoutPad + 127) / 128;
    const int mTiles   = (HLR * WLR) / MPB;               // 128

    int bid = blockIdx.x;
    const int n      = bid / (mTiles * ncBlocks);
    const int rem    = bid % (mTiles * ncBlocks);
    const int mTile  = rem / ncBlocks;
    const int ncTile = (rem % ncBlocks) * 8 + wave;       // 16-col tile index

    const int mBase = mTile * MPB;
    const int mLoc  = lane & 15;
    const int half  = lane >> 4;
    const bool active = ncTile < ncTiles;

    const bf16_t* xn = xbf + (size_t)n * (CIN * HLR * WLR);

    // staging role: thread owns (pixel sm = tid&31, 72 contiguous k = slot*72..)
    const int sm   = threadIdx.x & 31;
    const int slot = threadIdx.x >> 5;                    // 0..7
    const int spm  = mBase + sm;
    const int sy   = spm >> 6, sx = spm & 63;
    const int sklB = slot * 72;                           // 72 = 8 ci-groups of 9

    v8f acc0 = {}, acc1 = {};

    for (int kc = 0; kc < KTOT; kc += KCHUNK) {
        // ---- cooperative im2col stage: 32 pixels x KCHUNK into LDS ----
        {
            const int ci0 = (kc + sklB) / 9;              // 8 consecutive ci
            bf16_t* dst = &Als[sm * KSTRIDE + sklB];
            #pragma unroll
            for (int g = 0; g < 8; ++g) {
                const bf16_t* plane = xn + (ci0 + g) * (HLR * WLR);
                #pragma unroll
                for (int ky = 0; ky < 3; ++ky) {
                    const int iy = sy + ky - 1;
                    const bool rowOk = (unsigned)iy < (unsigned)HLR;
                    const bf16_t* row = plane + iy * WLR;
                    #pragma unroll
                    for (int kx = 0; kx < 3; ++kx) {
                        const int ix = sx + kx - 1;
                        bf16_t v = (bf16_t)0.f;
                        if (rowOk && (unsigned)ix < (unsigned)WLR) v = row[ix];
                        dst[g * 9 + ky * 3 + kx] = v;
                    }
                }
            }
        }
        __syncthreads();

        if (active) {
            // B fragments for this chunk: 18 k-tiles, contiguous per tile
            const bf16_t* bChunk = Bpack
                + ((size_t)(((kc >> 5) * ncTiles + ncTile) * 32 + lane) << 4);
            for (int kk = 0; kk < KCHUNK; kk += 32) {
                // B fragment: one 32-byte contiguous load per lane, reused 2x
                const bf16_t* bp = bChunk + (size_t)(kk >> 5) * (ncTiles * 32 * 16);
                v16bf b = *(const v16bf*)bp;
                __builtin_prefetch(bp + ncTiles * 32 * 16, 0, 1);

                // A fragments: two contiguous 8-element groups per lane
                const bf16_t* ap0 = &Als[mLoc * KSTRIDE + kk + half * 16];
                const bf16_t* ap1 = ap0 + 16 * KSTRIDE;
                v8bf a0lo = *(const v8bf*)(ap0);
                v8bf a0hi = *(const v8bf*)(ap0 + 8);
                v8bf a1lo = *(const v8bf*)(ap1);
                v8bf a1hi = *(const v8bf*)(ap1 + 8);
                v16bf a0, a1;
                #pragma unroll
                for (int t = 0; t < 8; ++t) {
                    a0[t] = a0lo[t]; a0[t + 8] = a0hi[t];
                    a1[t] = a1lo[t]; a1[t + 8] = a1hi[t];
                }

                acc0 = __builtin_amdgcn_wmma_f32_16x16x32_bf16(
                           false, a0, false, b, (short)0, acc0, false, false);
                acc1 = __builtin_amdgcn_wmma_f32_16x16x32_bf16(
                           false, a1, false, b, (short)0, acc1, false, false);
            }
        }
        __syncthreads();
    }

    if (!active) return;
    const int c = ncTile * 16 + (lane & 15);
    if (c >= CoutReal) return;
    const float bv = bias[c];
    const size_t chanBase = (size_t)(n * CoutReal + c) * (HLR * WLR);

    // D layout: lane holds 8 consecutive pixels m = mBase + s*16 + 8*half + r
    #pragma unroll
    for (int s = 0; s < 2; ++s) {
        const v8f& acc = s ? acc1 : acc0;
        const int m0 = mBase + s * 16 + 8 * half;
        if (outBf) {
            v8bf res;
            #pragma unroll
            for (int r = 0; r < 8; ++r) {
                float v = acc[r] + bv;
                if (relu) v = fmaxf(v, 0.f);
                res[r] = (bf16_t)v;
            }
            *(v8bf*)(out_bf + chanBase + m0) = res;
        } else {
            v8f res;
            #pragma unroll
            for (int r = 0; r < 8; ++r) {
                float v = acc[r] + bv;
                if (relu) v = fmaxf(v, 0.f);
                res[r] = v;
            }
            *(v8f*)(out_f32 + chanBase + m0) = res;
        }
    }
}

// ---------------------------------------------------------------------------
// Fused RAFT convex upsample + identity grid + affine.
// Writes flow (out slots 1 and 3) and delta_flow (slot 2).
// ---------------------------------------------------------------------------
__global__ void upsample_affine(const float* __restrict__ flow_lr, // (N,2,64,64)
                                const float* __restrict__ mask,    // (N,576,64,64)
                                const float* __restrict__ bw,      // (N,2,3)
                                float* __restrict__ d_out)
{
    int idx = blockIdx.x * blockDim.x + threadIdx.x;
    const int total = NBATCH * HF * WF;
    if (idx >= total) return;
    int wf = idx % WF;
    int hf = (idx / WF) % HF;
    int n  = idx / (WF * HF);
    int h = hf >> 3, u = hf & 7, w = wf >> 3, v = wf & 7;

    float mk[9], mmax = -1e30f;
    const float* mbase = mask + ((size_t)n * 576 + u * 8 + v) * (HLR * WLR) + h * WLR + w;
    #pragma unroll
    for (int k = 0; k < 9; ++k) {
        mk[k] = mbase[(size_t)k * 64 * (HLR * WLR)];
        mmax = fmaxf(mmax, mk[k]);
    }
    float s = 0.f;
    #pragma unroll
    for (int k = 0; k < 9; ++k) { mk[k] = __expf(mk[k] - mmax); s += mk[k]; }
    const float inv = 1.f / s;

    float dx = 0.f, dy = 0.f;
    const float* fxp = flow_lr + (size_t)(n * 2 + 0) * (HLR * WLR);
    const float* fyp = flow_lr + (size_t)(n * 2 + 1) * (HLR * WLR);
    #pragma unroll
    for (int k = 0; k < 9; ++k) {
        int iy = h + k / 3 - 1, ix = w + k % 3 - 1;
        float fx = 0.f, fy = 0.f;
        if ((unsigned)iy < (unsigned)HLR && (unsigned)ix < (unsigned)WLR) {
            int o = iy * WLR + ix;
            fx = (float)FD * fxp[o];
            fy = (float)FD * fyp[o];
        }
        float wk = mk[k] * inv;
        dx += wk * fx; dy += wk * fy;
    }

    float gx = (wf + 0.5f) * (2.f / WF) - 1.f + dx;
    float gy = (hf + 0.5f) * (2.f / HF) - 1.f + dy;
    const float* b = bw + n * 6;
    float ox = b[0] * gx + b[1] * gy + b[2];
    float oy = b[3] * gx + b[4] * gy + b[5];

    const size_t outImg = (size_t)NBATCH * 3 * HF * WF;
    const size_t flowSz = (size_t)NBATCH * HF * WF * 2;
    size_t p = ((size_t)(n * HF + hf) * WF + wf) * 2;
    d_out[outImg + p]                  = ox;  // flow
    d_out[outImg + p + 1]              = oy;
    d_out[outImg + flowSz + p]         = dx;  // delta_flow
    d_out[outImg + flowSz + p + 1]     = dy;
    d_out[outImg + 2 * flowSz + p]     = ox;  // flow (dup output)
    d_out[outImg + 2 * flowSz + p + 1] = oy;
}

// ---------------------------------------------------------------------------
// Bilinear grid_sample, padding_mode='border', align_corners=False.
// ---------------------------------------------------------------------------
__global__ void grid_sample_kernel(const float* __restrict__ img,  // (N,3,512,512)
                                   float* __restrict__ d_out)
{
    int idx = blockIdx.x * blockDim.x + threadIdx.x;
    const int total = NBATCH * 3 * HF * WF;
    if (idx >= total) return;
    int wf = idx % WF;
    int hf = (idx / WF) % HF;
    int c  = (idx / (WF * HF)) % 3;
    int n  = idx / (WF * HF * 3);

    const size_t outImg = (size_t)NBATCH * 3 * HF * WF;
    size_t p = ((size_t)(n * HF + hf) * WF + wf) * 2;
    float ox = d_out[outImg + p];
    float oy = d_out[outImg + p + 1];

    float x = fminf(fmaxf((ox + 1.f) * (WF * 0.5f) - 0.5f, 0.f), (float)(WF - 1));
    float y = fminf(fmaxf((oy + 1.f) * (HF * 0.5f) - 0.5f, 0.f), (float)(HF - 1));
    float x0 = floorf(x), y0 = floorf(y);
    float wx = x - x0, wy = y - y0;
    int x0i = (int)x0, y0i = (int)y0;
    int x1i = min(x0i + 1, WF - 1), y1i = min(y0i + 1, HF - 1);

    const float* im = img + (size_t)(n * 3 + c) * (HF * WF);
    float g00 = im[y0i * WF + x0i];
    float g01 = im[y0i * WF + x1i];
    float g10 = im[y1i * WF + x0i];
    float g11 = im[y1i * WF + x1i];
    d_out[idx] = g00 * (1.f - wx) * (1.f - wy) + g01 * wx * (1.f - wy)
               + g10 * (1.f - wx) * wy + g11 * wx * wy;
}

// ---------------------------------------------------------------------------
extern "C" void kernel_launch(void* const* d_in, const int* in_sizes, int n_in,
                              void* d_out_v, int out_size, void* d_ws, size_t ws_size,
                              hipStream_t stream)
{
    const float* img       = (const float*)d_in[0];
    const float* features  = (const float*)d_in[1];
    const float* base_warp = (const float*)d_in[2];
    const float* w_f1 = (const float*)d_in[3];
    const float* b_f1 = (const float*)d_in[4];
    const float* w_f2 = (const float*)d_in[5];
    const float* b_f2 = (const float*)d_in[6];
    const float* w_m1 = (const float*)d_in[7];
    const float* b_m1 = (const float*)d_in[8];
    const float* w_m2 = (const float*)d_in[9];
    const float* b_m2 = (const float*)d_in[10];
    float* d_out = (float*)d_out_v;

    char* ws = (char*)d_ws;
    size_t off = 0;
    auto alloc = [&](size_t bytes) -> void* {
        void* p = ws + off;
        off = (off + bytes + 255) & ~(size_t)255;
        return p;
    };
    bf16_t* featBf = (bf16_t*)alloc((size_t)NBATCH * CIN * HLR * WLR * 2);
    bf16_t* hBf    = (bf16_t*)alloc((size_t)NBATCH * CIN * HLR * WLR * 2);
    bf16_t* hmBf   = (bf16_t*)alloc((size_t)NBATCH * CIN * HLR * WLR * 2);
    bf16_t* Bf1    = (bf16_t*)alloc((size_t)KTOT * 256 * 2);
    bf16_t* Bm1    = (bf16_t*)alloc((size_t)KTOT * 256 * 2);
    bf16_t* Bm2    = (bf16_t*)alloc((size_t)KTOT * 576 * 2);
    bf16_t* Bf2    = (bf16_t*)alloc((size_t)KTOT * 16 * 2);
    float*  flowLr = (float*)alloc((size_t)NBATCH * 2 * HLR * WLR * 4);
    float*  maskB  = (float*)alloc((size_t)NBATCH * 576 * HLR * WLR * 4);

    const float scale = 1.f / sqrtf((float)(CIN * 9));   // EqualConv 1/sqrt(fan_in)

    pack_weights_frag<<<(KTOT * 256 + 255) / 256, 256, 0, stream>>>(w_f1, Bf1, 256, 256, scale);
    pack_weights_frag<<<(KTOT * 256 + 255) / 256, 256, 0, stream>>>(w_m1, Bm1, 256, 256, scale);
    pack_weights_frag<<<(KTOT * 576 + 255) / 256, 256, 0, stream>>>(w_m2, Bm2, 576, 576, scale);
    pack_weights_frag<<<(KTOT * 16  + 255) / 256, 256, 0, stream>>>(w_f2, Bf2,   2,  16, scale);

    const int nFeat = NBATCH * CIN * HLR * WLR;
    f32_to_bf16<<<(nFeat + 255) / 256, 256, 0, stream>>>(features, featBf, nFeat);

    const int mTiles = (HLR * WLR) / MPB;                // 128
    // conv f1: features -> h (relu, bf16 out)
    conv3x3_wmma<<<NBATCH * mTiles * 2, 256, 0, stream>>>(featBf, Bf1, b_f1,
                                                          nullptr, hBf, 256, 256, 1, 1);
    // conv m1: features -> hm (relu, bf16 out)
    conv3x3_wmma<<<NBATCH * mTiles * 2, 256, 0, stream>>>(featBf, Bm1, b_m1,
                                                          nullptr, hmBf, 256, 256, 1, 1);
    // conv f2: h -> flow_lr (2 real channels, padded to 16; fp32 out)
    conv3x3_wmma<<<NBATCH * mTiles * 1, 256, 0, stream>>>(hBf, Bf2, b_f2,
                                                          flowLr, nullptr, 16, 2, 0, 0);
    // conv m2: hm -> mask (576 channels; fp32 out)
    conv3x3_wmma<<<NBATCH * mTiles * 5, 256, 0, stream>>>(hmBf, Bm2, b_m2,
                                                          maskB, nullptr, 576, 576, 0, 0);

    upsample_affine<<<(NBATCH * HF * WF + 255) / 256, 256, 0, stream>>>(
        flowLr, maskB, base_warp, d_out);
    grid_sample_kernel<<<(NBATCH * 3 * HF * WF + 255) / 256, 256, 0, stream>>>(
        img, d_out);
}
